// CenterAttention_55843164783270
// MI455X (gfx1250) — compile-verified, matrix operand order checked by reading, not run on hardware
//
#include <hip/hip_runtime.h>

typedef __attribute__((ext_vector_type(16))) __bf16 v16bf;
typedef __attribute__((ext_vector_type(8)))  __bf16 v8bf;
typedef __attribute__((ext_vector_type(8)))  float  v8f;

#define B_     256
#define N_     64
#define D_     512
#define H_     8
#define INNER_ 512
#define SCALE_ 0.125f

// ---- LDS layout. All row strides are multiples of 8 bf16 (16 B) so each
// WMMA fragment loads as two ds_load_b128; all OFF_* are 16B-aligned. ----
#define XT_LD  72    // XT[512][72]  bf16 : x^T  (d-major, n contiguous)
#define WT_LD  72    // WspT[128][72] bf16: Wspec^T cols 0..127 (k contiguous);
                     //   q-columns (j<64) pre-scaled by SCALE_
#define QK_LD  136   // QK[512][136] bf16 : cols 0..63 q_s*scale | 64..127 k_s
constexpr int OFF_XT   = 0;
constexpr int OFF_WSPT = OFF_XT   + 512 * XT_LD * 2;   // 73728
constexpr int OFF_QK   = OFF_WSPT + 128 * WT_LD * 2;   // 92160
constexpr int OFF_QC   = OFF_QK   + 512 * QK_LD * 2;   // 231424
constexpr int OFF_WT   = OFF_QC   + 512 * 4;
constexpr int OFF_A    = OFF_WT   + 8 * 512 * 4;
constexpr int OFF_Y    = OFF_A    + 512 * 4;
constexpr int OFF_OUTV = OFF_Y    + 8 * 512 * 4;
constexpr int OFF_S    = OFF_OUTV + 512 * 4;
constexpr int OFF_T    = OFF_S    + 512 * 4;
constexpr int OFF_TW   = OFF_T    + 512 * 4;
constexpr int LDS_BYTES = OFF_TW  + 8 * 512 * 4;       // 290816 B < 320 KB

// One WMMA 16-bit operand fragment = two contiguous 8-bf16 runs per lane
// (K = base+8g..+7 in VGPRs 0-3, K = base+16+8g..+7 in VGPRs 4-7)
// -> two ds_load_b128 from a K-contiguous, 16B-aligned row.
__device__ __forceinline__ v16bf ldfrag(const __bf16* p) {
  union { v16bf v; v8bf h[2]; } u;
  u.h[0] = *(const v8bf*)(p);
  u.h[1] = *(const v8bf*)(p + 16);
  return u.v;
}

__device__ __forceinline__ float rmax16(float v) {
  #pragma unroll
  for (int m = 1; m < 16; m <<= 1) v = fmaxf(v, __shfl_xor(v, m, 32));
  return v;
}
__device__ __forceinline__ float rsum16(float v) {
  #pragma unroll
  for (int m = 1; m < 16; m <<= 1) v += __shfl_xor(v, m, 32);
  return v;
}

__global__ __launch_bounds__(256)
void center_attn_fused_kernel(const float* __restrict__ x,
                              const float* __restrict__ Wq,
                              const float* __restrict__ Wkv,
                              const float* __restrict__ Wout,
                              const float* __restrict__ bout,
                              const float* __restrict__ Wspec,
                              float* __restrict__ out) {
  extern __shared__ char smem[];
  __bf16* XT   = (__bf16*)(smem + OFF_XT);
  __bf16* WspT = (__bf16*)(smem + OFF_WSPT);
  __bf16* QK   = (__bf16*)(smem + OFF_QK);
  float* qc   = (float*)(smem + OFF_QC);
  float* wt   = (float*)(smem + OFF_WT);
  float* aL   = (float*)(smem + OFF_A);
  float* yv   = (float*)(smem + OFF_Y);
  float* outv = (float*)(smem + OFF_OUTV);
  float* sv   = (float*)(smem + OFF_S);
  float* tv   = (float*)(smem + OFF_T);
  float* tw   = (float*)(smem + OFF_TW);

  const int b    = blockIdx.x;
  const int tid  = threadIdx.x;
  const int lane = tid & 31;
  const int w    = tid >> 5;
  const float* xb = x + (size_t)b * N_ * D_;

  // ---- Stage 1: coalesced f32 loads; transpose-convert into bf16 LDS.
  for (int i = tid; i < (N_ * D_) / 4; i += 256) {
    float4 v = ((const float4*)xb)[i];
    int flat = i * 4;
    int n = flat >> 9, d = flat & 511;
    XT[(d + 0) * XT_LD + n] = (__bf16)v.x;
    XT[(d + 1) * XT_LD + n] = (__bf16)v.y;
    XT[(d + 2) * XT_LD + n] = (__bf16)v.z;
    XT[(d + 3) * XT_LD + n] = (__bf16)v.w;
  }
  for (int i = tid; i < 64 * 128; i += 256) {
    int k = i >> 7, j = i & 127;                  // coalesced in j
    float wv = Wspec[k * 192 + j];
    WspT[j * WT_LD + k] = (__bf16)(j < 64 ? wv * SCALE_ : wv); // fold scale into q_s
  }
  __syncthreads();

  // ---- Stage 2 (spatial, folded; all n-broadcast): qc = x[b,32,:] @ Wq
  #pragma unroll
  for (int kk = 0; kk < 2; ++kk) {
    int j = tid + kk * 256;
    float acc = 0.f;
    for (int d = 0; d < D_; ++d)
      acc += (float)XT[d * XT_LD + 32] * Wq[d * INNER_ + j];
    qc[j] = acc;
  }
  __syncthreads();

  // wt[h][d] = sum_j Wk[d, h*64+j] * qc[h*64+j]
  #pragma unroll
  for (int kk = 0; kk < 16; ++kk) {
    int idx = tid + kk * 256, h = idx >> 9, d = idx & 511;
    float acc = 0.f;
    const float* wkc = Wkv + (size_t)d * 1024 + h * 64;
    const float* q   = qc + h * 64;
    for (int j = 0; j < 64; ++j) acc += wkc[j] * q[j];
    wt[idx] = acc;
  }
  __syncthreads();

  // scores[h][m] = scale * sum_d X[m][d] wt[h][d]; softmax over m (8x64)
  #pragma unroll
  for (int kk = 0; kk < 2; ++kk) {
    int idx = tid + kk * 256, h = idx >> 6, m = idx & 63;
    float acc = 0.f;
    const float* wh = wt + h * 512;
    for (int d = 0; d < D_; ++d) acc += (float)XT[d * XT_LD + m] * wh[d];
    aL[idx] = acc * SCALE_;
  }
  __syncthreads();
  if (tid < 8) {
    float mx = -1e30f;
    for (int m = 0; m < 64; ++m) mx = fmaxf(mx, aL[tid * 64 + m]);
    float Z = 0.f;
    for (int m = 0; m < 64; ++m) { float e = __expf(aL[tid * 64 + m] - mx); aL[tid * 64 + m] = e; Z += e; }
    float inv = 1.f / Z;
    for (int m = 0; m < 64; ++m) aL[tid * 64 + m] *= inv;
  }
  __syncthreads();

  // y[h][d] = sum_m attn[h][m] * X[m][d]  (m contiguous in XT row)
  #pragma unroll
  for (int kk = 0; kk < 16; ++kk) {
    int idx = tid + kk * 256, h = idx >> 9, d = idx & 511;
    float acc = 0.f;
    const float* ah = aL + h * 64;
    const __bf16* xr = XT + d * XT_LD;
    for (int m = 0; m < 64; ++m) acc += ah[m] * (float)xr[m];
    yv[idx] = acc;
  }
  __syncthreads();

  // outv[h*64+j] = sum_d y[h][d] * Wv[d, h*64+j]
  #pragma unroll
  for (int kk = 0; kk < 2; ++kk) {
    int idx = tid + kk * 256, h = idx >> 6, j = idx & 63;
    float acc = 0.f;
    const float* yh = yv + h * 512;
    const float* wv = Wkv + 512 + h * 64 + j;
    for (int d = 0; d < 512; ++d) acc += yh[d] * wv[(size_t)d * 1024];
    outv[idx] = acc;
  }
  __syncthreads();

  // s[e] = bout[e] + outv . Wout[:,e]; zero per-wave accumulators
  #pragma unroll
  for (int kk = 0; kk < 2; ++kk) {
    int e = tid + kk * 256;
    float acc = bout[e];
    for (int j = 0; j < 512; ++j) acc += outv[j] * Wout[j * 512 + e];
    sv[e] = acc;
  }
  for (int i = tid; i < 8 * 512; i += 256) tw[i] = 0.f;
  __syncthreads();

  // ---- Stage 3: GEMM1 (WMMA bf16): QK(512x128) = X^T(512x64) @ Wspec(64x128)
  {
    const int mloc = lane & 15, g = lane >> 4;
    for (int tile = w; tile < 256; tile += 8) {
      int mt = tile >> 3, nt = tile & 7;
      v8f acc = {};
      #pragma unroll
      for (int kt = 0; kt < 2; ++kt) {
        v16bf af = ldfrag(XT   + (mt * 16 + mloc) * XT_LD + kt * 32 + 8 * g);
        v16bf bf = ldfrag(WspT + (nt * 16 + mloc) * WT_LD + kt * 32 + 8 * g);
        acc = __builtin_amdgcn_wmma_f32_16x16x32_bf16(false, af, false, bf,
                                                      (short)0, acc, false, false);
      }
      #pragma unroll
      for (int r = 0; r < 8; ++r)
        QK[(mt * 16 + r + 8 * g) * QK_LD + nt * 16 + mloc] = (__bf16)acc[r];
    }
  }
  __syncthreads();

  // ---- Stage 4: t[e] = sum_d s_d * softmax_row(q_s k_s^T)[e]
  // Three WMMA sweeps with per-lane-only stats (cross-lane reduce once/strip):
  //   sweep 1: row max   sweep 2: row sum-exp   sweep 3: weighted accumulate
  {
    const int nloc = lane & 15, g = lane >> 4;
    for (int strip = w; strip < 32; strip += 8) {
      const int row0 = strip * 16;
      v16bf afr[2];
      #pragma unroll
      for (int kt = 0; kt < 2; ++kt)
        afr[kt] = ldfrag(QK + (row0 + nloc) * QK_LD + kt * 32 + 8 * g);

      // sweep 1: per-lane running max (z == acc; scale folded into q_s)
      float mp[8];
      #pragma unroll
      for (int r = 0; r < 8; ++r) mp[r] = -1e30f;
      for (int nt = 0; nt < 32; ++nt) {
        v8f acc = {};
        #pragma unroll
        for (int kt = 0; kt < 2; ++kt) {
          v16bf bfr = ldfrag(QK + (nt * 16 + nloc) * QK_LD + 64 + kt * 32 + 8 * g);
          acc = __builtin_amdgcn_wmma_f32_16x16x32_bf16(false, afr[kt], false, bfr,
                                                        (short)0, acc, false, false);
        }
        #pragma unroll
        for (int r = 0; r < 8; ++r) mp[r] = fmaxf(mp[r], acc[r]);
      }
      float mrow[8];
      #pragma unroll
      for (int r = 0; r < 8; ++r) mrow[r] = rmax16(mp[r]);

      // sweep 2: per-lane partial sum of exp
      float zp[8];
      #pragma unroll
      for (int r = 0; r < 8; ++r) zp[r] = 0.f;
      for (int nt = 0; nt < 32; ++nt) {
        v8f acc = {};
        #pragma unroll
        for (int kt = 0; kt < 2; ++kt) {
          v16bf bfr = ldfrag(QK + (nt * 16 + nloc) * QK_LD + 64 + kt * 32 + 8 * g);
          acc = __builtin_amdgcn_wmma_f32_16x16x32_bf16(false, afr[kt], false, bfr,
                                                        (short)0, acc, false, false);
        }
        #pragma unroll
        for (int r = 0; r < 8; ++r) zp[r] += __expf(acc[r] - mrow[r]);
      }
      float crow[8];
      #pragma unroll
      for (int r = 0; r < 8; ++r)
        crow[r] = sv[row0 + r + 8 * g] / rsum16(zp[r]);

      // sweep 3: t_e += sum_r c_r * exp(z - m_r)
      float* twp = tw + w * 512;
      for (int nt = 0; nt < 32; ++nt) {
        v8f acc = {};
        #pragma unroll
        for (int kt = 0; kt < 2; ++kt) {
          v16bf bfr = ldfrag(QK + (nt * 16 + nloc) * QK_LD + 64 + kt * 32 + 8 * g);
          acc = __builtin_amdgcn_wmma_f32_16x16x32_bf16(false, afr[kt], false, bfr,
                                                        (short)0, acc, false, false);
        }
        float colsum = 0.f;
        #pragma unroll
        for (int r = 0; r < 8; ++r)
          colsum += crow[r] * __expf(acc[r] - mrow[r]);
        colsum += __shfl_xor(colsum, 16, 32);      // rows 0-7 + rows 8-15
        if (g == 0) twp[nt * 16 + nloc] += colsum; // wave-private, race-free
      }
    }
  }
  __syncthreads();

  // ---- Stage 5: reduce per-wave accumulators; broadcast over n to output
  #pragma unroll
  for (int kk = 0; kk < 2; ++kk) {
    int e = tid + kk * 256;
    float acc = 0.f;
    #pragma unroll
    for (int ww = 0; ww < 8; ++ww) acc += tw[ww * 512 + e];
    tv[e] = acc;
  }
  __syncthreads();
  float* ob = out + (size_t)b * N_ * D_;
  for (int i = tid; i < (N_ * D_) / 4; i += 256) {
    int flat = i * 4, e = flat & 511;
    float4 v;
    v.x = tv[e]; v.y = tv[e + 1]; v.z = tv[e + 2]; v.w = tv[e + 3];
    ((float4*)ob)[i] = v;
  }
}

extern "C" void kernel_launch(void* const* d_in, const int* in_sizes, int n_in,
                              void* d_out, int out_size, void* d_ws, size_t ws_size,
                              hipStream_t stream) {
  const float* x     = (const float*)d_in[0];
  const float* Wq    = (const float*)d_in[1];
  const float* Wkv   = (const float*)d_in[2];
  const float* Wout  = (const float*)d_in[3];
  const float* bout  = (const float*)d_in[4];
  const float* Wspec = (const float*)d_in[5];
  float* out = (float*)d_out;
  (void)in_sizes; (void)n_in; (void)out_size; (void)d_ws; (void)ws_size;

  center_attn_fused_kernel<<<B_, 256, LDS_BYTES, stream>>>(
      x, Wq, Wkv, Wout, bout, Wspec, out);
}
